// MultiHeadAttention_26577257628374
// MI455X (gfx1250) — compile-verified, matrix operand order checked by reading, not run on hardware
//
#include <hip/hip_runtime.h>

// ---------------------------------------------------------------------------
// MultiHeadAttention + residual + LayerNorm for MI455X (gfx1250, wave32, WMMA)
// B=8, S=1024, D=1024, H=16, dk=64
// d_out = [attn (8*16*1024*1024 f32)] ++ [out (8*1024*1024 f32)]
// GEMMs: v_wmma_f32_16x16x32_bf16, f32 accumulate.
// Staging: global_load_async_to_lds_b128 + double-buffered LDS when available.
// ---------------------------------------------------------------------------

typedef __attribute__((ext_vector_type(16))) __bf16 v16bf;
typedef __attribute__((ext_vector_type(8)))  __bf16 bf16x8;
typedef __attribute__((ext_vector_type(8)))  float  v8f;
typedef int v4i_ __attribute__((vector_size(16)));   // matches builtin's V4i

#define B_   8
#define S_   1024
#define D_   1024
#define H_   16
#define DK_  64
#define LDA_ 40   // LDS row stride (bf16 elems) for 32-wide K tiles, 16B aligned
#define LDB_ 40

#if defined(__gfx1250__) && __has_builtin(__builtin_amdgcn_global_load_async_to_lds_b128)
#define ASYNC_LDS 1
#else
#define ASYNC_LDS 0
#endif

__device__ __forceinline__ void cp16(const __bf16* g, __bf16* l) {
#if ASYNC_LDS
  __builtin_amdgcn_global_load_async_to_lds_b128(
      (__attribute__((address_space(1))) v4i_*)g,
      (__attribute__((address_space(3))) v4i_*)l, 0, 0);
#else
  *(uint4*)l = *(const uint4*)g;
#endif
}

__device__ __forceinline__ void async_wait_all() {
#if ASYNC_LDS
#if __has_builtin(__builtin_amdgcn_s_wait_asynccnt)
  __builtin_amdgcn_s_wait_asynccnt(0);
#else
  asm volatile("s_wait_asynccnt 0x0" ::: "memory");
#endif
#endif
}

__device__ __forceinline__ __bf16 f2bf(float f) {
  union { float f; unsigned u; } in; in.f = f;
  unsigned u = in.u;
  u += 0x7FFFu + ((u >> 16) & 1u);          // round-to-nearest-even
  union { unsigned short s; __bf16 b; } out;
  out.s = (unsigned short)(u >> 16);
  return out.b;
}

// ------------------------- f32 -> bf16 convert -----------------------------
__global__ __launch_bounds__(256) void cvt_bf16_kernel(const float* __restrict__ in,
                                                       __bf16* __restrict__ out, int n) {
  int i = (blockIdx.x * 256 + threadIdx.x) * 4;
  if (i < n) {
    float4 f = *(const float4*)&in[i];
    __align__(8) __bf16 t[4] = { f2bf(f.x), f2bf(f.y), f2bf(f.z), f2bf(f.w) };
    *(uint2*)&out[i] = *(const uint2*)t;
  }
}

// ------------------ fragment loaders (LDS, documented layout) --------------
// A 16x32 bf16 frag: lane row m = L%16; K halves selected by L/16.
// Elements 0..7  -> K =  8*half + e
// Elements 8..15 -> K = 16 + 8*half + (e-8)
// B stored as [n][k] in LDS so B frags load with the identical pattern.
__device__ __forceinline__ v16bf load_frag(const __bf16* row, int half) {
  union { v16bf v; bf16x8 h[2]; } u;
  u.h[0] = *(const bf16x8*)(row + 8 * half);
  u.h[1] = *(const bf16x8*)(row + 16 + 8 * half);
  return u.v;
}

// --------------------- projection GEMM: C = A @ W^T ------------------------
// A [8192,1024] bf16 row-major, W [1024,1024] bf16 row-major (torch [out,in]),
// C [8192,1024] bf16.  Block tile 128(M) x 64(N), 8 waves, K-step 32,
// double-buffered LDS with async global->LDS copies.
__global__ __launch_bounds__(256) void proj_gemm_kernel(const __bf16* __restrict__ A,
                                                        const __bf16* __restrict__ W,
                                                        __bf16* __restrict__ C) {
  __shared__ __align__(16) __bf16 As[2][128 * LDA_];
  __shared__ __align__(16) __bf16 Bs[2][64 * LDB_];
  const int tid  = threadIdx.x;
  const int m0   = blockIdx.y * 128;
  const int n0   = blockIdx.x * 64;
  const int wave = tid >> 5, lane = tid & 31;
  const int lm   = lane & 15, lh = lane >> 4;

  auto stage = [&](int buf, int k0) {
    #pragma unroll
    for (int i = 0; i < 2; ++i) {                  // A tile 128x32: 512 slots
      int slot = tid + i * 256, r = slot >> 2, cg = slot & 3;
      cp16(&A[(size_t)(m0 + r) * D_ + k0 + cg * 8], &As[buf][r * LDA_ + cg * 8]);
    }
    { int r = tid >> 2, cg = tid & 3;              // B tile 64x32 as [n][k]
      cp16(&W[(size_t)(n0 + r) * D_ + k0 + cg * 8], &Bs[buf][r * LDB_ + cg * 8]); }
  };

  v8f acc[4] = {};
  stage(0, 0);
  async_wait_all();
  __syncthreads();
  for (int k0 = 0; k0 < D_; k0 += 32) {
    const int cur = (k0 >> 5) & 1;
    if (k0 + 32 < D_) stage(cur ^ 1, k0 + 32);     // prefetch next K-step

    v16bf a = load_frag(&As[cur][(wave * 16 + lm) * LDA_], lh);
    #pragma unroll
    for (int nt = 0; nt < 4; ++nt) {
      v16bf b = load_frag(&Bs[cur][(nt * 16 + lm) * LDB_], lh);
      acc[nt] = __builtin_amdgcn_wmma_f32_16x16x32_bf16(false, a, false, b,
                                                        (short)0, acc[nt], false, false);
    }
    async_wait_all();
    __syncthreads();
  }
  // C/D layout: VGPR r, lane: M = 8*(L/16)+r, N = L%16
  #pragma unroll
  for (int nt = 0; nt < 4; ++nt)
    #pragma unroll
    for (int r = 0; r < 8; ++r) {
      int gm = m0 + wave * 16 + lh * 8 + r;
      int gn = n0 + nt * 16 + lm;
      C[(size_t)gm * D_ + gn] = f2bf(acc[nt][r]);
    }
}

// -------------------- scores = Q K^T / sqrt(dk) per (b,h) ------------------
// grid: (S/128, S/64, B*H).  K-dim = dk = 64 (two 32-steps, double-buffered).
__global__ __launch_bounds__(256) void scores_gemm_kernel(const __bf16* __restrict__ Q,
                                                          const __bf16* __restrict__ K,
                                                          float* __restrict__ attn) {
  __shared__ __align__(16) __bf16 As[2][128 * LDA_];
  __shared__ __align__(16) __bf16 Bs[2][64 * LDB_];
  const int tid = threadIdx.x;
  const int bh  = blockIdx.z, b = bh >> 4, h = bh & 15;
  const int q0  = blockIdx.x * 128;
  const int kk0 = blockIdx.y * 64;
  const int wave = tid >> 5, lane = tid & 31;
  const int lm = lane & 15, lh = lane >> 4;
  const int colbase = h * DK_;

  auto stage = [&](int buf, int k0) {
    #pragma unroll
    for (int i = 0; i < 2; ++i) {
      int slot = tid + i * 256, r = slot >> 2, cg = slot & 3;
      cp16(&Q[(size_t)(b * S_ + q0 + r) * D_ + colbase + k0 + cg * 8],
           &As[buf][r * LDA_ + cg * 8]);
    }
    { int r = tid >> 2, cg = tid & 3;
      cp16(&K[(size_t)(b * S_ + kk0 + r) * D_ + colbase + k0 + cg * 8],
           &Bs[buf][r * LDB_ + cg * 8]); }
  };

  v8f acc[4] = {};
  stage(0, 0);
  async_wait_all();
  __syncthreads();
  for (int k0 = 0; k0 < DK_; k0 += 32) {
    const int cur = (k0 >> 5) & 1;
    if (k0 + 32 < DK_) stage(cur ^ 1, k0 + 32);

    v16bf a = load_frag(&As[cur][(wave * 16 + lm) * LDA_], lh);
    #pragma unroll
    for (int nt = 0; nt < 4; ++nt) {
      v16bf bb = load_frag(&Bs[cur][(nt * 16 + lm) * LDB_], lh);
      acc[nt] = __builtin_amdgcn_wmma_f32_16x16x32_bf16(false, a, false, bb,
                                                        (short)0, acc[nt], false, false);
    }
    async_wait_all();
    __syncthreads();
  }
  const float scale = 0.125f;  // 1/sqrt(64)
  #pragma unroll
  for (int nt = 0; nt < 4; ++nt)
    #pragma unroll
    for (int r = 0; r < 8; ++r) {
      int gq = q0 + wave * 16 + lh * 8 + r;
      int gk = kk0 + nt * 16 + lm;
      attn[((size_t)bh * S_ + gq) * S_ + gk] = acc[nt][r] * scale;
    }
}

// -------------------- in-place masked softmax over keys --------------------
// one block per (b,h,q) row of 1024
__global__ __launch_bounds__(256) void softmax_kernel(float* __restrict__ attn,
                                                      const unsigned char* __restrict__ mask) {
  __shared__ float red[256];
  const int row = blockIdx.x;          // b*H*S + h*S + q
  const int b   = row >> 14;           // / (H_*S_)
  float* srow = attn + (size_t)row * S_;
  const unsigned char* mrow = mask + (size_t)b * S_;
  const int tid = threadIdx.x;

  float v[4];
  float lmax = -3.0e38f;
  #pragma unroll
  for (int j = 0; j < 4; ++j) {
    int c = j * 256 + tid;
    float s = srow[c];
    if (!mrow[c]) s = -9.0e15f;
    v[j] = s;
    lmax = fmaxf(lmax, s);
  }
  red[tid] = lmax; __syncthreads();
  for (int s = 128; s > 0; s >>= 1) { if (tid < s) red[tid] = fmaxf(red[tid], red[tid + s]); __syncthreads(); }
  const float m = red[0]; __syncthreads();

  float lsum = 0.f;
  #pragma unroll
  for (int j = 0; j < 4; ++j) { v[j] = __expf(v[j] - m); lsum += v[j]; }
  red[tid] = lsum; __syncthreads();
  for (int s = 128; s > 0; s >>= 1) { if (tid < s) red[tid] += red[tid + s]; __syncthreads(); }
  const float inv = 1.0f / red[0];
  #pragma unroll
  for (int j = 0; j < 4; ++j) srow[j * 256 + tid] = v[j] * inv;
}

// -------------------- ctx = attn @ V per (b,h) -----------------------------
// grid: (S/128, B*H). N = dk = 64 whole head, K loop over 1024 keys.
// attn f32 converted to bf16 while staging; V staged transposed [d][kk].
__global__ __launch_bounds__(256) void av_gemm_kernel(const float* __restrict__ attn,
                                                      const __bf16* __restrict__ V,
                                                      float* __restrict__ ctx) {
  __shared__ __align__(16) __bf16 As[128 * LDA_];
  __shared__ __align__(16) __bf16 Bs[64 * LDB_];
  const int tid = threadIdx.x;
  const int bh  = blockIdx.y, b = bh >> 4, h = bh & 15;
  const int q0  = blockIdx.x * 128;
  const int wave = tid >> 5, lane = tid & 31;
  const int lm = lane & 15, lh = lane >> 4;

  v8f acc[4] = {};
  for (int kk0 = 0; kk0 < S_; kk0 += 32) {
    // A: attn[128 x 32] f32 -> bf16 (1024 float4 slots, 4/thread)
    #pragma unroll
    for (int i = 0; i < 4; ++i) {
      int slot = tid + i * 256, r = slot >> 3, cg = slot & 7;
      float4 f = *(const float4*)&attn[((size_t)bh * S_ + q0 + r) * S_ + kk0 + cg * 4];
      __bf16* dst = &As[r * LDA_ + cg * 4];
      dst[0] = f2bf(f.x); dst[1] = f2bf(f.y); dst[2] = f2bf(f.z); dst[3] = f2bf(f.w);
    }
    // B: V[32 kk x 64 d] staged transposed -> Bs[d][kk] (8 elems/thread)
    { int kr = tid & 31, dbase = (tid >> 5) * 8;
      const __bf16* vr = &V[(size_t)(b * S_ + kk0 + kr) * D_ + h * DK_ + dbase];
      #pragma unroll
      for (int j = 0; j < 8; ++j) Bs[(dbase + j) * LDB_ + kr] = vr[j]; }
    __syncthreads();

    v16bf a = load_frag(&As[(wave * 16 + lm) * LDA_], lh);
    #pragma unroll
    for (int nt = 0; nt < 4; ++nt) {
      v16bf bb = load_frag(&Bs[(nt * 16 + lm) * LDB_], lh);
      acc[nt] = __builtin_amdgcn_wmma_f32_16x16x32_bf16(false, a, false, bb,
                                                        (short)0, acc[nt], false, false);
    }
    __syncthreads();
  }
  #pragma unroll
  for (int nt = 0; nt < 4; ++nt)
    #pragma unroll
    for (int r = 0; r < 8; ++r) {
      int gq = q0 + wave * 16 + lh * 8 + r;
      int gd = h * DK_ + nt * 16 + lm;
      ctx[(size_t)(b * S_ + gq) * D_ + gd] = acc[nt][r];
    }
}

// -------------------- residual + LayerNorm ---------------------------------
__global__ __launch_bounds__(256) void layernorm_kernel(const float* __restrict__ ctx,
                                                        const float* __restrict__ x,
                                                        const float* __restrict__ gamma,
                                                        const float* __restrict__ beta,
                                                        float* __restrict__ out) {
  __shared__ float red[256], red2[256];
  const int row = blockIdx.x;          // b*S + s
  const float* cr = ctx + (size_t)row * D_;
  const float* xr = x   + (size_t)row * D_;
  float* orow = out + (size_t)row * D_;
  const int tid = threadIdx.x;

  float v[4], lsum = 0.f, lsq = 0.f;
  #pragma unroll
  for (int j = 0; j < 4; ++j) {
    int c = j * 256 + tid;
    float t = cr[c] + xr[c];
    v[j] = t; lsum += t; lsq += t * t;
  }
  red[tid] = lsum; red2[tid] = lsq; __syncthreads();
  for (int s = 128; s > 0; s >>= 1) {
    if (tid < s) { red[tid] += red[tid + s]; red2[tid] += red2[tid + s]; }
    __syncthreads();
  }
  const float mu  = red[0] * (1.0f / D_);
  const float var = red2[0] * (1.0f / D_) - mu * mu;
  const float rs  = rsqrtf(var + 1e-6f);
  #pragma unroll
  for (int j = 0; j < 4; ++j) {
    int c = j * 256 + tid;
    orow[c] = (v[j] - mu) * rs * gamma[c] + beta[c];
  }
}

// ---------------------------------------------------------------------------
extern "C" void kernel_launch(void* const* d_in, const int* in_sizes, int n_in,
                              void* d_out, int out_size, void* d_ws, size_t ws_size,
                              hipStream_t stream) {
  (void)in_sizes; (void)n_in; (void)out_size; (void)ws_size;
  const float* x    = (const float*)d_in[0];
  const unsigned char* mask = (const unsigned char*)d_in[1];   // jax bool -> 1 byte
  const float* Wq   = (const float*)d_in[2];
  const float* Wk   = (const float*)d_in[3];
  const float* Wv   = (const float*)d_in[4];
  const float* gamma = (const float*)d_in[5];
  const float* beta  = (const float*)d_in[6];

  float* attn = (float*)d_out;                                   // [B,H,S,S]
  float* outp = attn + (size_t)B_ * H_ * S_ * S_;                // [B,S,D]

  char* ws = (char*)d_ws;
  __bf16* Qh  = (__bf16*)(ws);                                   // 16 MB
  __bf16* Kh  = (__bf16*)(ws + ((size_t)16 << 20));              // 16 MB
  __bf16* Vh  = (__bf16*)(ws + ((size_t)32 << 20));              // 16 MB
  __bf16* Xh  = (__bf16*)(ws + ((size_t)48 << 20));              // 16 MB
  __bf16* Wqh = (__bf16*)(ws + ((size_t)64 << 20));              //  2 MB
  __bf16* Wkh = (__bf16*)(ws + ((size_t)66 << 20));              //  2 MB
  __bf16* Wvh = (__bf16*)(ws + ((size_t)68 << 20));              //  2 MB
  float*  Ctx = (float*)(ws);   // aliases Qh/Kh (dead after scores GEMM), 32 MB

  const int nX = B_ * S_ * D_;       // 8388608
  const int nW = D_ * D_;            // 1048576

  cvt_bf16_kernel<<<nX / 1024, 256, 0, stream>>>(x,  Xh,  nX);
  cvt_bf16_kernel<<<nW / 1024, 256, 0, stream>>>(Wq, Wqh, nW);
  cvt_bf16_kernel<<<nW / 1024, 256, 0, stream>>>(Wk, Wkh, nW);
  cvt_bf16_kernel<<<nW / 1024, 256, 0, stream>>>(Wv, Wvh, nW);

  dim3 gproj(D_ / 64, (B_ * S_) / 128);                          // (16, 64)
  proj_gemm_kernel<<<gproj, 256, 0, stream>>>(Xh, Wqh, Qh);
  proj_gemm_kernel<<<gproj, 256, 0, stream>>>(Xh, Wkh, Kh);
  proj_gemm_kernel<<<gproj, 256, 0, stream>>>(Xh, Wvh, Vh);

  dim3 gsc(S_ / 128, S_ / 64, B_ * H_);                          // (8, 16, 128)
  scores_gemm_kernel<<<gsc, 256, 0, stream>>>(Qh, Kh, attn);

  softmax_kernel<<<B_ * H_ * S_, 256, 0, stream>>>(attn, mask);  // 131072 rows

  dim3 gav(S_ / 128, B_ * H_);                                   // (8, 128)
  av_gemm_kernel<<<gav, 256, 0, stream>>>(attn, Vh, Ctx);

  layernorm_kernel<<<B_ * S_, 256, 0, stream>>>(Ctx, x, gamma, beta, outp);
}
